// RotationModulation_81484119540141
// MI455X (gfx1250) — matrix operand order repaired and checked by reading, not compiled, not run on hardware
//
#include <hip/hip_runtime.h>
#include <cstdint>
#include <cstddef>

#define NFEAT 1152
#define DSUB  64
#define NSKEW 2016
#define BATCH 16
#define TLEN  1024

typedef __bf16 v16bf __attribute__((ext_vector_type(16)));
typedef __bf16 v8bf  __attribute__((ext_vector_type(8)));
typedef float  v8f   __attribute__((ext_vector_type(8)));

__device__ __forceinline__ __bf16 bf(float f) { return (__bf16)f; }

__device__ __forceinline__ v16bf cat8(v8bf lo, v8bf hi) {
  return __builtin_shufflevector(lo, hi, 0,1,2,3,4,5,6,7,8,9,10,11,12,13,14,15);
}

// ---------------------------------------------------------------------------
// Stage 1a: Gram matrix G = P^T P (64x64)
// ---------------------------------------------------------------------------
__global__ void k_gram(const float* __restrict__ P, float* __restrict__ G) {
  int i = blockIdx.x, j = threadIdx.x;
  float s = 0.f;
  for (int r = 0; r < NFEAT; ++r) s += P[r*DSUB + i] * P[r*DSUB + j];
  G[i*DSUB + j] = s;
}

// ---------------------------------------------------------------------------
// Stage 1b: Cholesky-QR. L = chol(G); U = P L^-T. Writes U (bf16, NxD) and
// U^T (bf16, DxN).
// ---------------------------------------------------------------------------
__global__ void k_cholqr(const float* __restrict__ P, const float* __restrict__ G,
                         unsigned short* __restrict__ Ub_,
                         unsigned short* __restrict__ Ut_) {
  __bf16* Ub = (__bf16*)Ub_;
  __bf16* Ut = (__bf16*)Ut_;
  __shared__ float M[DSUB*DSUB];
  const int tid = threadIdx.x;  // 64 threads
  for (int j = 0; j < DSUB; ++j) M[tid*DSUB + j] = G[tid*DSUB + j];
  __syncthreads();
  for (int k = 0; k < DSUB; ++k) {
    if (tid == 0) M[k*DSUB + k] = sqrtf(M[k*DSUB + k]);
    __syncthreads();
    if (tid > k) M[tid*DSUB + k] /= M[k*DSUB + k];
    __syncthreads();
    if (tid > k) {
      float lik = M[tid*DSUB + k];
      for (int j = k + 1; j <= tid; ++j) M[tid*DSUB + j] -= lik * M[j*DSUB + k];
    }
    __syncthreads();
  }
  // per-row forward substitution: u L^T = p  <=>  L u = p (triangular solve)
  for (int base = 0; base < NFEAT; base += DSUB) {
    int r = base + tid;
    float u[DSUB];
    for (int j = 0; j < DSUB; ++j) {
      float s = P[r*DSUB + j];
      for (int k = 0; k < j; ++k) s -= M[j*DSUB + k] * u[k];
      u[j] = s / M[j*DSUB + j];
    }
    for (int j = 0; j < DSUB; ++j) {
      __bf16 h = bf(u[j]);
      Ub[r*DSUB + j]  = h;
      Ut[j*NFEAT + r] = h;
    }
  }
}

// ---------------------------------------------------------------------------
// Stage 2a: scT[c][b] = silu(cond[b][c])  (transposed for GEMV locality)
// ---------------------------------------------------------------------------
__global__ void k_silu(const float* __restrict__ cond, float* __restrict__ scT) {
  int e = blockIdx.x * blockDim.x + threadIdx.x;  // 72*256 == 1152*16
  int c = e >> 4, b = e & 15;
  float xv = cond[b*NFEAT + c];
  scT[c*16 + b] = xv / (1.f + __expf(-xv));
}

// ---------------------------------------------------------------------------
// Stage 2b: v[b][k] = dot(silu(cond[b]), W[k]) + bias[k]; one wave per k row.
// ---------------------------------------------------------------------------
__global__ void k_v(const float* __restrict__ W, const float* __restrict__ bias,
                    const float* __restrict__ scT, float* __restrict__ v) {
  const int lane = threadIdx.x & 31, wave = threadIdx.x >> 5;
  const int k = blockIdx.x * 8 + wave;  // grid 252 * 8 waves == 2016 exactly
  float acc[BATCH];
  #pragma unroll
  for (int b = 0; b < BATCH; ++b) acc[b] = 0.f;
  for (int c = lane; c < NFEAT; c += 32) {
    float wv = W[k*NFEAT + c];
    const float* s = &scT[c*16];
    #pragma unroll
    for (int b = 0; b < BATCH; ++b) acc[b] += wv * s[b];
  }
  #pragma unroll
  for (int b = 0; b < BATCH; ++b) {
    float a = acc[b];
    for (int m = 16; m; m >>= 1) a += __shfl_xor(a, m, 32);
    if (lane == 0) v[b*NSKEW + k] = a + bias[k];
  }
}

// ---------------------------------------------------------------------------
// Stage 3: per-batch expm of 64x64 skew matrix (scaling-and-squaring Taylor).
// Writes Gt[b][n][k] = (expm(A)-I)[k][n] as bf16 (B-fragment friendly layout).
// ---------------------------------------------------------------------------
__global__ void k_expm(const float* __restrict__ v, unsigned short* __restrict__ Gt_) {
  __bf16* Gt = (__bf16*)Gt_;
  __shared__ float A[4096], T[4096], Tmp[4096];
  __shared__ float rs[DSUB];
  __shared__ int   s_sh;
  __shared__ float sc_sh;
  const int tid = threadIdx.x, b = blockIdx.x;
  const float* vb = v + b*NSKEW;
  for (int e = tid; e < 4096; e += 256) {
    int i = e >> 6, j = e & 63;
    float val = 0.f;
    if (i < j)      val =  vb[(i*(127 - i))/2 + (j - i - 1)];
    else if (i > j) val = -vb[(j*(127 - j))/2 + (i - j - 1)];
    A[e] = val;
  }
  __syncthreads();
  if (tid < DSUB) {
    float s = 0.f;
    for (int j = 0; j < DSUB; ++j) s += fabsf(A[tid*DSUB + j]);
    rs[tid] = s;
  }
  __syncthreads();
  if (tid == 0) {
    float m = 0.f;
    for (int i = 0; i < DSUB; ++i) m = fmaxf(m, rs[i]);
    int s = 0;
    while (m > 0.25f && s < 40) { m *= 0.5f; ++s; }
    s_sh = s;
    sc_sh = ldexpf(1.0f, -s);
  }
  __syncthreads();
  const float scl = sc_sh;
  const int   sq  = s_sh;
  for (int e = tid; e < 4096; e += 256) A[e] *= scl;
  for (int e = tid; e < 4096; e += 256) T[e] = ((e >> 6) == (e & 63)) ? 1.f : 0.f;
  __syncthreads();
  for (int it = 12; it >= 1; --it) {
    for (int e = tid; e < 4096; e += 256) {
      int i = e >> 6, j = e & 63;
      float s = 0.f;
      for (int k = 0; k < DSUB; ++k) s += A[i*DSUB + k] * T[k*DSUB + j];
      Tmp[e] = s;
    }
    __syncthreads();
    float inv = 1.0f / (float)it;
    for (int e = tid; e < 4096; e += 256)
      T[e] = (((e >> 6) == (e & 63)) ? 1.f : 0.f) + Tmp[e] * inv;
    __syncthreads();
  }
  for (int t = 0; t < sq; ++t) {
    for (int e = tid; e < 4096; e += 256) {
      int i = e >> 6, j = e & 63;
      float s = 0.f;
      for (int k = 0; k < DSUB; ++k) s += T[i*DSUB + k] * T[k*DSUB + j];
      Tmp[e] = s;
    }
    __syncthreads();
    for (int e = tid; e < 4096; e += 256) T[e] = Tmp[e];
    __syncthreads();
  }
  for (int e = tid; e < 4096; e += 256) {
    int n = e >> 6, k = e & 63;
    float g = T[k*DSUB + n] - ((k == n) ? 1.f : 0.f);
    Gt[b*4096 + e] = bf(g);
  }
}

// ---------------------------------------------------------------------------
// Stage 4 (hot): out = x + ((x@U) @ G) @ U^T, fused, bf16 WMMA, f32 accum.
// Block = 128 threads (4 waves), 64 rows per block, grid = (16, 16 batches).
// Fragments come straight from global memory (no hot-loop LDS/barriers).
// B-fragment loads are hoisted ahead of the A conversion so the in-order
// LOADcnt waits decay incrementally instead of draining before each WMMA;
// K-loop unrolled 2x for deeper load/compute overlap.
// ---------------------------------------------------------------------------
__global__ void __launch_bounds__(128)
k_main(const float* __restrict__ x, const unsigned short* __restrict__ Ub_,
       const unsigned short* __restrict__ Ut_, const unsigned short* __restrict__ Gt_,
       float* __restrict__ out) {
  const __bf16* Ub = (const __bf16*)Ub_;
  const __bf16* Ut = (const __bf16*)Ut_;
  const __bf16* Gt = (const __bf16*)Gt_;

  __shared__ alignas(64) __bf16 lds_z[64*64];   // 8 KB  z tile (bf16)
  __shared__ alignas(64) __bf16 lds_g[64*64];   // 8 KB  G^T [n][k]
  __shared__ alignas(64) __bf16 lds_w[64*64];   // 8 KB  w tile (bf16)

  const int tid  = threadIdx.x;
  const int lane = tid & 31;
  const int ln   = lane & 15;   // row/col within 16-wide tile
  const int lh   = lane >> 4;   // lane half
  const int wave = tid >> 5;    // 4 waves == 4 M-strips of 16 rows
  const int batch = blockIdx.y;
  const int row0  = blockIdx.x * 64;
  const int xrow  = batch*TLEN + row0;
  const int klo   = lh * 16;    // A-fragment K sub-offset (f32 elements)

  v8f zero8 = {0.f,0.f,0.f,0.f,0.f,0.f,0.f,0.f};
  v8f acc[4];
  acc[0] = zero8; acc[1] = zero8; acc[2] = zero8; acc[3] = zero8;

  // ---------------- Phase A: z = x_tile @ U (K = 1152, 36 steps) -----------
  const float* xp = x + (size_t)(xrow + wave*16 + ln)*NFEAT;   // this lane's row
  const __bf16* up = Ut + (size_t)ln*NFEAT + lh*16;            // this lane's U^T rows
  #pragma unroll 2
  for (int kb = 0; kb < NFEAT/32; ++kb) {
    const int base = kb*32;
    // issue the x row-segment loads first ...
    float4 f0 = *(const float4*)(xp + base + klo);
    float4 f1 = *(const float4*)(xp + base + klo + 4);
    float4 f2 = *(const float4*)(xp + base + 8 + klo);
    float4 f3 = *(const float4*)(xp + base + 8 + klo + 4);
    // ... then all four B-fragments, so waits can drain incrementally
    v16bf bbv[4];
    #pragma unroll
    for (int nt = 0; nt < 4; ++nt)
      bbv[nt] = *(const v16bf*)(up + (size_t)(nt*16)*NFEAT + base);
    v16bf a;
    a[0]=bf(f0.x); a[1]=bf(f0.y); a[2]=bf(f0.z); a[3]=bf(f0.w);
    a[4]=bf(f1.x); a[5]=bf(f1.y); a[6]=bf(f1.z); a[7]=bf(f1.w);
    a[8]=bf(f2.x); a[9]=bf(f2.y); a[10]=bf(f2.z); a[11]=bf(f2.w);
    a[12]=bf(f3.x); a[13]=bf(f3.y); a[14]=bf(f3.z); a[15]=bf(f3.w);
    #pragma unroll
    for (int nt = 0; nt < 4; ++nt)
      acc[nt] = __builtin_amdgcn_wmma_f32_16x16x32_bf16(false, a, false, bbv[nt],
                                                        (short)0, acc[nt], false, false);
  }

  // store z as bf16 (C-layout: element i -> row i+8*lh, col ln); load G^T
  #pragma unroll
  for (int nt = 0; nt < 4; ++nt)
    #pragma unroll
    for (int i = 0; i < 8; ++i)
      lds_z[(wave*16 + i + 8*lh)*64 + nt*16 + ln] = bf(acc[nt][i]);
  #pragma unroll
  for (int q = 0; q < 4; ++q)
    *(uint4*)(lds_g + tid*32 + q*8) = *(const uint4*)(Gt + batch*4096 + tid*32 + q*8);
  __syncthreads();

  // ---------------- Phase B: w = z @ G (K = 64) ----------------------------
  v8f wacc[4];
  wacc[0] = zero8; wacc[1] = zero8; wacc[2] = zero8; wacc[3] = zero8;
  {
    const int zrow = wave*16 + ln;
    #pragma unroll
    for (int ks = 0; ks < 2; ++ks) {
      v8bf alo = *(const v8bf*)(lds_z + zrow*64 + ks*32 + klo);
      v8bf ahi = *(const v8bf*)(lds_z + zrow*64 + ks*32 + 8 + klo);
      v16bf a = cat8(alo, ahi);
      #pragma unroll
      for (int nt = 0; nt < 4; ++nt) {
        v16bf bb = *(const v16bf*)(lds_g + (nt*16 + ln)*64 + ks*32 + lh*16);
        wacc[nt] = __builtin_amdgcn_wmma_f32_16x16x32_bf16(false, a, false, bb,
                                                           (short)0, wacc[nt], false, false);
      }
    }
  }
  #pragma unroll
  for (int nt = 0; nt < 4; ++nt)
    #pragma unroll
    for (int i = 0; i < 8; ++i)
      lds_w[(wave*16 + i + 8*lh)*64 + nt*16 + ln] = bf(wacc[nt][i]);
  __syncthreads();

  // hoist w A-fragments (stable for all of phase C)
  v16bf aw0, aw1;
  {
    const int wrow = wave*16 + ln;
    v8bf l0 = *(const v8bf*)(lds_w + wrow*64 + klo);
    v8bf h0 = *(const v8bf*)(lds_w + wrow*64 + 8 + klo);
    v8bf l1 = *(const v8bf*)(lds_w + wrow*64 + 32 + klo);
    v8bf h1 = *(const v8bf*)(lds_w + wrow*64 + 40 + klo);
    aw0 = cat8(l0, h0);
    aw1 = cat8(l1, h1);
  }

  // ---------------- Phase C: out = x + w @ U^T (72 N-tiles) ----------------
  const __bf16* ubp = Ub + (size_t)ln*DSUB + lh*16;
  #pragma unroll 4
  for (int nt = 0; nt < NFEAT/16; ++nt) {
    v16bf b0 = *(const v16bf*)(ubp + (size_t)(nt*16)*DSUB);
    v16bf b1 = *(const v16bf*)(ubp + (size_t)(nt*16)*DSUB + 32);
    v8f c = zero8;
    c = __builtin_amdgcn_wmma_f32_16x16x32_bf16(false, aw0, false, b0, (short)0, c, false, false);
    c = __builtin_amdgcn_wmma_f32_16x16x32_bf16(false, aw1, false, b1, (short)0, c, false, false);
    const size_t gbase = (size_t)(xrow + wave*16 + 8*lh)*NFEAT + nt*16 + ln;
    #pragma unroll
    for (int i = 0; i < 8; ++i) {
      const size_t gi = gbase + (size_t)i*NFEAT;
      out[gi] = x[gi] + c[i];
    }
  }
}

// ---------------------------------------------------------------------------
// Launcher
// ---------------------------------------------------------------------------
extern "C" void kernel_launch(void* const* d_in, const int* in_sizes, int n_in,
                              void* d_out, int out_size, void* d_ws, size_t ws_size,
                              hipStream_t stream) {
  (void)in_sizes; (void)n_in; (void)out_size; (void)ws_size;
  const float* x    = (const float*)d_in[0];
  const float* cond = (const float*)d_in[1];
  const float* P    = (const float*)d_in[2];
  const float* W    = (const float*)d_in[3];
  const float* bias = (const float*)d_in[4];
  float* out = (float*)d_out;
  char* ws = (char*)d_ws;

  float*          G64  = (float*)(ws + 0);                 // 16384 B
  unsigned short* Ubf  = (unsigned short*)(ws + 16384);    // 147456 B
  unsigned short* Utbf = (unsigned short*)(ws + 163840);   // 147456 B
  float*          scT  = (float*)(ws + 311296);            // 73728 B
  float*          v    = (float*)(ws + 385024);            // 129024 B
  unsigned short* Gt   = (unsigned short*)(ws + 514048);   // 131072 B

  k_gram  <<<64, 64, 0, stream>>>(P, G64);
  k_cholqr<<<1, 64, 0, stream>>>(P, G64, Ubf, Utbf);
  k_silu  <<<72, 256, 0, stream>>>(cond, scT);
  k_v     <<<252, 256, 0, stream>>>(W, bias, scT, v);
  k_expm  <<<16, 256, 0, stream>>>(v, Gt);
  dim3 grid(16, 16);
  k_main  <<<grid, 128, 0, stream>>>(x, Ubf, Utbf, Gt, out);
}
